// TransformerConvE_23218593202853
// MI455X (gfx1250) — matrix operand order repaired and checked by reading, not compiled, not run on hardware
//
#include <hip/hip_runtime.h>
#include <hip/hip_bf16.h>

typedef __attribute__((ext_vector_type(16))) _Float16 v16h;
typedef __attribute__((ext_vector_type(8)))  float    v8f;

#define LRELU_SLOPE 0.15f
#define INV_SQRT_C  0.17677669529663687f   // 1/sqrt(32)

// ---------------------------------------------------------------- helpers

__device__ inline void atomAddF32(float* p, float v) {
  // lowers to global_atomic_add_f32 (no-return) at agent scope
  __hip_atomic_fetch_add(p, v, __ATOMIC_RELAXED, __HIP_MEMORY_SCOPE_AGENT);
}

__device__ inline void atomMaxF32(float* addr, float val) {
  // IEEE trick: max over floats via signed-int max (val>=0) / unsigned min (val<0)
  if (val >= 0.0f) atomicMax((int*)addr, __float_as_int(val));
  else             atomicMin((unsigned int*)addr, __float_as_uint(val));
}

// A fragment (16x32 f16) from fp32 row-major source, row stride 128
__device__ inline v16h load_a_f32(const float* __restrict__ rowbase, int k0, int hi) {
  const float* p = rowbase + k0 + hi * 16;
  v16h a;
#pragma unroll
  for (int j = 0; j < 8; ++j) { a[j] = (_Float16)p[j]; a[8 + j] = (_Float16)p[8 + j]; }
  return a;
}

// A fragment from f16 LDS tile [16][128]
__device__ inline v16h load_a_lds(const _Float16* __restrict__ rowbase, int k0, int hi) {
  const _Float16* p = rowbase + k0 + hi * 16;
  v16h a;
#pragma unroll
  for (int j = 0; j < 8; ++j) { a[j] = p[j]; a[8 + j] = p[8 + j]; }
  return a;
}

// B fragment (32x16 f16): transposed f16 weights Wt[out_col][128], one 32B load
__device__ inline v16h load_b(const _Float16* __restrict__ Wt, int ncol, int k0, int hi) {
  return *(const v16h*)(Wt + (size_t)ncol * 128 + k0 + hi * 16);
}

// ---------------------------------------------------------------- kernels

// Convert weights to f16, transposed [out_col][k].
// Wnode_t covers Wq|Wk|Wv|Ws (512 cols), We2t and Wedt are 128 cols each.
__global__ __launch_bounds__(256) void prep_weights_kernel(
    const float* __restrict__ Wq, const float* __restrict__ Wk,
    const float* __restrict__ Wv, const float* __restrict__ Ws,
    const float* __restrict__ We2, const float* __restrict__ Wed,
    _Float16* __restrict__ Wnode_t, _Float16* __restrict__ We2t,
    _Float16* __restrict__ Wedt) {
  int idx = blockIdx.x * 256 + threadIdx.x;          // 0 .. 98303
  if (idx < 65536) {                                 // 512 x 128
    int n = idx >> 7, k = idx & 127;
    const float* W = (n < 128) ? Wq : (n < 256) ? Wk : (n < 384) ? Wv : Ws;
    Wnode_t[idx] = (_Float16)W[k * 128 + (n & 127)];
  } else {
    int t = idx - 65536;                             // 0 .. 32767
    int n = t >> 7, k = t & 127;
    if (n < 128) We2t[n * 128 + k] = (_Float16)We2[k * 128 + n];
    else         Wedt[(n - 128) * 128 + k] = (_Float16)Wed[k * 128 + (n - 128)];
  }
}

__global__ __launch_bounds__(256) void init_kernel(
    float* __restrict__ out, float* __restrict__ amax,
    float* __restrict__ denom, int N) {
  long long idx = (long long)blockIdx.x * 256 + threadIdx.x;
  long long n128 = (long long)N * 128, n4 = (long long)N * 4;
  if (idx < n128)                out[idx] = 0.0f;
  else if (idx < n128 + n4)      amax[idx - n128] = -__builtin_inff();
  else if (idx < n128 + 2 * n4)  denom[idx - n128 - n4] = 0.0f;
}

// Fused node projections: [q|k|v] (f16, L2-resident for the edge gathers)
// and x_skip (fp32) = x @ [Wq|Wk|Wv|Ws] + bias.
// Block = 8 waves = one 16-row tile of x across all 512 output columns.
__global__ __launch_bounds__(256) void node_proj_kernel(
    const float* __restrict__ x, const _Float16* __restrict__ Wnode_t,
    const float* __restrict__ bq, const float* __restrict__ bk,
    const float* __restrict__ bv, const float* __restrict__ bs,
    _Float16* __restrict__ qkvh, float* __restrict__ xr, int N) {
  const int wave = threadIdx.x >> 5;
  const int lane = threadIdx.x & 31;
  const int r = lane & 15, hi = lane >> 4;
  const int row0 = blockIdx.x * 16;
  const float* xrow = x + (size_t)(row0 + r) * 128;

  v8f acc[4] = {};
#pragma unroll
  for (int kc = 0; kc < 4; ++kc) {
    const int k0 = kc * 32;
    v16h a = load_a_f32(xrow, k0, hi);               // reused across 4 tiles
#pragma unroll
    for (int t = 0; t < 4; ++t) {
      v16h b = load_b(Wnode_t, wave * 64 + t * 16 + r, k0, hi);
      acc[t] = __builtin_amdgcn_wmma_f32_16x16x32_f16(
          false, a, false, b, (short)0, acc[t], false, false);
    }
  }
#pragma unroll
  for (int t = 0; t < 4; ++t) {
    const int gcol = wave * 64 + t * 16 + r;         // 0..511
    const int m = gcol >> 7, c = gcol & 127;         // m uniform per tile
    if (m < 3) {
      const float* bias = (m == 0) ? bq : (m == 1) ? bk : bv;
      const float bval = bias[c];
      _Float16* dstp = qkvh + (size_t)m * N * 128;
#pragma unroll
      for (int g = 0; g < 8; ++g)
        dstp[(size_t)(row0 + g + hi * 8) * 128 + c] = (_Float16)(acc[t][g] + bval);
    } else {
      const float bval = bs[c];
#pragma unroll
      for (int g = 0; g < 8; ++g)
        xr[(size_t)(row0 + g + hi * 8) * 128 + c] = acc[t][g] + bval;
    }
  }
}

// Edge pipeline: t = lrelu(ea@We1+be1); e = t@We2+be2 (LDS); ed = e@Wed+bed (f16).
// Block = 16 edges, 8 waves each own a 16-col N-tile.
__global__ __launch_bounds__(256) void edge_pipeline_kernel(
    const float* __restrict__ edge_attr,
    const float* __restrict__ We1, const float* __restrict__ be1,
    const _Float16* __restrict__ We2t, const float* __restrict__ be2,
    const _Float16* __restrict__ Wedt, const float* __restrict__ bed,
    _Float16* __restrict__ ed) {
  __shared__ _Float16 tA[16][128];
  __shared__ _Float16 tB[16][128];
  const long long e0 = (long long)blockIdx.x * 16;
  const int tid = threadIdx.x;

  // stage 1: K=7 MLP + leaky relu, straight VALU
#pragma unroll
  for (int i = 0; i < 8; ++i) {
    int idx = tid + i * 256;
    int row = idx >> 7, col = idx & 127;
    const float* ea = edge_attr + (size_t)(e0 + row) * 7;
    float s = be1[col];
#pragma unroll
    for (int kk = 0; kk < 7; ++kk) s += ea[kk] * We1[kk * 128 + col];
    s = (s > 0.0f) ? s : LRELU_SLOPE * s;
    tA[row][col] = (_Float16)s;
  }
  __syncthreads();

  const int wave = tid >> 5, lane = tid & 31;
  const int r = lane & 15, hi = lane >> 4;
  const int n0 = wave * 16;

  // stage 2: e = tA @ We2 + be2  -> tB (f16)
  v8f acc = {};
#pragma unroll
  for (int kc = 0; kc < 4; ++kc) {
    const int k0 = kc * 32;
    v16h a = load_a_lds(&tA[r][0], k0, hi);
    v16h b = load_b(We2t, n0 + r, k0, hi);
    acc = __builtin_amdgcn_wmma_f32_16x16x32_f16(
        false, a, false, b, (short)0, acc, false, false);
  }
  {
    const float bval = be2[n0 + r];
#pragma unroll
    for (int g = 0; g < 8; ++g) tB[g + hi * 8][n0 + r] = (_Float16)(acc[g] + bval);
  }
  __syncthreads();

  // stage 3: ed = e @ Wed + bed -> global f16
  v8f acc2 = {};
#pragma unroll
  for (int kc = 0; kc < 4; ++kc) {
    const int k0 = kc * 32;
    v16h a = load_a_lds(&tB[r][0], k0, hi);
    v16h b = load_b(Wedt, n0 + r, k0, hi);
    acc2 = __builtin_amdgcn_wmma_f32_16x16x32_f16(
        false, a, false, b, (short)0, acc2, false, false);
  }
  {
    const float bval = bed[n0 + r];
#pragma unroll
    for (int g = 0; g < 8; ++g)
      ed[(size_t)(e0 + g + hi * 8) * 128 + n0 + r] = (_Float16)(acc2[g] + bval);
  }
}

// Wave per edge: alpha[e,h] = q[dst]·(k[src]+ed[e]) / sqrt(C); segment max.
__global__ __launch_bounds__(256) void alpha_kernel(
    const int* __restrict__ ei, const _Float16* __restrict__ q,
    const _Float16* __restrict__ k, const _Float16* __restrict__ ed,
    float* __restrict__ alpha, float* __restrict__ amax, long long E) {
  const long long e = (long long)blockIdx.x * 8 + (threadIdx.x >> 5);
  if (e >= E) return;
  const int lane = threadIdx.x & 31;
  const int src = ei[e], dst = ei[E + e];
  const _Float16* qd  = q  + (size_t)dst * 128;
  const _Float16* ks  = k  + (size_t)src * 128;
  const _Float16* edr = ed + (size_t)e * 128;
  float s[4];
#pragma unroll
  for (int h = 0; h < 4; ++h) {
    const int ch = h * 32 + lane;
    s[h] = (float)qd[ch] * ((float)ks[ch] + (float)edr[ch]);
  }
#pragma unroll
  for (int m = 16; m >= 1; m >>= 1)
#pragma unroll
    for (int h = 0; h < 4; ++h) s[h] += __shfl_xor(s[h], m, 32);
  if (lane < 4) {
    const float a = s[lane] * INV_SQRT_C;
    alpha[e * 4 + lane] = a;
    atomMaxF32(&amax[(size_t)dst * 4 + lane], a);
  }
}

// Per (edge, head): exp(alpha - max) and segment-sum denominator.
__global__ __launch_bounds__(256) void expsum_kernel(
    const int* __restrict__ ei, float* __restrict__ alpha,
    const float* __restrict__ amax, float* __restrict__ denom, long long E) {
  const long long idx = (long long)blockIdx.x * 256 + threadIdx.x;
  if (idx >= E * 4) return;
  const long long e = idx >> 2;
  const int h = idx & 3;
  const int dst = ei[E + e];
  const float ea = __expf(alpha[idx] - amax[(size_t)dst * 4 + h]);
  alpha[idx] = ea;
  atomAddF32(&denom[(size_t)dst * 4 + h], ea);
}

// Wave per edge: out[dst] += (v[src]+ed) * alpha/denom
__global__ __launch_bounds__(256) void scatter_kernel(
    const int* __restrict__ ei, const _Float16* __restrict__ v,
    const _Float16* __restrict__ ed, const float* __restrict__ alpha,
    const float* __restrict__ denom, float* __restrict__ out, long long E) {
  const long long e = (long long)blockIdx.x * 8 + (threadIdx.x >> 5);
  if (e >= E) return;
  const int lane = threadIdx.x & 31;
  const int src = ei[e], dst = ei[E + e];
  const _Float16* vs  = v  + (size_t)src * 128;
  const _Float16* edr = ed + (size_t)e * 128;
#pragma unroll
  for (int h = 0; h < 4; ++h) {
    const float w = alpha[e * 4 + h] / denom[(size_t)dst * 4 + h];
    const int ch = h * 32 + lane;
    atomAddF32(&out[(size_t)dst * 128 + ch], ((float)vs[ch] + (float)edr[ch]) * w);
  }
}

// Wave per node: beta = sigmoid(out·(w1+w3) + xr·(w2-w3)); blend in place.
__global__ __launch_bounds__(256) void gate_kernel(
    const float* __restrict__ xr, const float* __restrict__ Wbeta,
    float* __restrict__ out, int N) {
  const int node = blockIdx.x * 8 + (threadIdx.x >> 5);
  if (node >= N) return;
  const int lane = threadIdx.x & 31;
  const float* o = out + (size_t)node * 128;
  const float* xn = xr + (size_t)node * 128;
  float ov[4], xv[4], dotp = 0.0f;
#pragma unroll
  for (int i = 0; i < 4; ++i) {
    const int ch = lane + i * 32;
    ov[i] = o[ch];
    xv[i] = xn[ch];
    const float w1 = Wbeta[ch], w2 = Wbeta[128 + ch], w3 = Wbeta[256 + ch];
    dotp += ov[i] * (w1 + w3) + xv[i] * (w2 - w3);
  }
#pragma unroll
  for (int m = 16; m >= 1; m >>= 1) dotp += __shfl_xor(dotp, m, 32);
  const float beta = 1.0f / (1.0f + __expf(-dotp));
#pragma unroll
  for (int i = 0; i < 4; ++i) {
    const int ch = lane + i * 32;
    out[(size_t)node * 128 + ch] = beta * xv[i] + (1.0f - beta) * ov[i];
  }
}

// ---------------------------------------------------------------- launch

extern "C" void kernel_launch(void* const* d_in, const int* in_sizes, int n_in,
                              void* d_out, int out_size, void* d_ws, size_t ws_size,
                              hipStream_t stream) {
  const float* x         = (const float*)d_in[0];
  const int*   ei        = (const int*)d_in[1];
  const float* edge_attr = (const float*)d_in[2];
  const float* We1 = (const float*)d_in[3];
  const float* be1 = (const float*)d_in[4];
  const float* We2 = (const float*)d_in[5];
  const float* be2 = (const float*)d_in[6];
  const float* Wq  = (const float*)d_in[7];
  const float* bq  = (const float*)d_in[8];
  const float* Wk  = (const float*)d_in[9];
  const float* bk  = (const float*)d_in[10];
  const float* Wv  = (const float*)d_in[11];
  const float* bv  = (const float*)d_in[12];
  const float* Wed = (const float*)d_in[13];
  const float* bed = (const float*)d_in[14];
  const float* Ws  = (const float*)d_in[15];
  const float* bs  = (const float*)d_in[16];
  const float* Wbeta = (const float*)d_in[17];

  const int       N = in_sizes[0] / 128;
  const long long E = (long long)in_sizes[2] / 7;

  // workspace carve (256B aligned)
  char* p = (char*)d_ws;
  auto carve = [&](size_t bytes) -> void* {
    void* r = (void*)p;
    p += (bytes + 255) & ~(size_t)255;
    return r;
  };
  _Float16* Wnode_t = (_Float16*)carve((size_t)512 * 128 * 2);
  _Float16* We2t    = (_Float16*)carve((size_t)128 * 128 * 2);
  _Float16* Wedt    = (_Float16*)carve((size_t)128 * 128 * 2);
  _Float16* qkvh    = (_Float16*)carve((size_t)3 * N * 128 * 2);  // q|k|v f16
  float*    xrbuf   = (float*)carve((size_t)N * 128 * sizeof(float));
  _Float16* edbuf   = (_Float16*)carve((size_t)E * 128 * 2);
  float*    alphab  = (float*)carve((size_t)E * 4 * sizeof(float));
  float*    amax    = (float*)carve((size_t)N * 4 * sizeof(float));
  float*    denom   = (float*)carve((size_t)N * 4 * sizeof(float));
  float*    outp    = (float*)d_out;

  prep_weights_kernel<<<(98304 + 255) / 256, 256, 0, stream>>>(
      Wq, Wk, Wv, Ws, We2, Wed, Wnode_t, We2t, Wedt);

  {
    const long long tot = (long long)N * 136;
    init_kernel<<<(int)((tot + 255) / 256), 256, 0, stream>>>(outp, amax, denom, N);
  }

  node_proj_kernel<<<N / 16, 256, 0, stream>>>(x, Wnode_t, bq, bk, bv, bs, qkvh, xrbuf, N);

  edge_pipeline_kernel<<<(int)(E / 16), 256, 0, stream>>>(
      edge_attr, We1, be1, We2t, be2, Wedt, bed, edbuf);

  const _Float16* qb = qkvh;
  const _Float16* kb = qkvh + (size_t)N * 128;
  const _Float16* vb = qkvh + (size_t)2 * N * 128;

  alpha_kernel<<<(int)((E + 7) / 8), 256, 0, stream>>>(ei, qb, kb, edbuf, alphab, amax, E);

  expsum_kernel<<<(int)((E * 4 + 255) / 256), 256, 0, stream>>>(ei, alphab, amax, denom, E);

  scatter_kernel<<<(int)((E + 7) / 8), 256, 0, stream>>>(ei, vb, edbuf, alphab, denom, outp, E);

  gate_kernel<<<(N + 7) / 8, 256, 0, stream>>>(xrbuf, Wbeta, outp, N);
}
